// VQ_16243566313849
// MI455X (gfx1250) — matrix operand order repaired and checked by reading, not compiled, not run on hardware
//
#include <hip/hip_runtime.h>
#include <hip/hip_bf16.h>

typedef __attribute__((ext_vector_type(16))) __bf16 v16bf;
typedef __attribute__((ext_vector_type(8)))  __bf16 v8bf;
typedef __attribute__((ext_vector_type(8)))  float  v8f;
typedef __attribute__((ext_vector_type(4)))  float  v4f;

#define B_   8
#define CIN  768
#define Q_   128
#define K_   4096
#define N_   4096
#define NB   128      // output columns per workgroup
#define ZTS  40       // padded LDS stride for z-transpose chunk (16B align, spreads banks)

#if defined(__has_builtin) && __has_builtin(__builtin_amdgcn_sched_barrier)
#define SCHED_FENCE() __builtin_amdgcn_sched_barrier(0)
#else
#define SCHED_FENCE() ((void)0)
#endif

union V16U { v16bf v; v8bf h[2]; };

// ---------------------------------------------------------------------------
// Prep: W -> bf16, emb -> bf16, emb_sq[k] = sum_q emb[k,q]^2   (all into ws)
// ---------------------------------------------------------------------------
__global__ __launch_bounds__(256) void vq_prep_kernel(
    const float* __restrict__ W, const float* __restrict__ emb,
    __bf16* __restrict__ Wbf, __bf16* __restrict__ embbf,
    float* __restrict__ embsq)
{
  int tid = blockIdx.x * blockDim.x + threadIdx.x;
  int nth = gridDim.x * blockDim.x;
  for (int i = tid; i < Q_ * CIN; i += nth) Wbf[i]   = (__bf16)W[i];
  for (int i = tid; i < K_ * Q_;  i += nth) embbf[i] = (__bf16)emb[i];
  for (int k = tid; k < K_;       k += nth) {
    const float* row = emb + (size_t)k * Q_;
    float s = 0.f;
    #pragma unroll 8
    for (int q = 0; q < Q_; ++q) s += row[q] * row[q];
    embsq[k] = s;
  }
}

// ---------------------------------------------------------------------------
// Fused: ze = W*z (WMMA bf16) -> argmin_k(embsq[k] - 2*emb·ze) -> gather emb
// One workgroup = (batch b, 128-column block of N). 8 wave32s.
// ---------------------------------------------------------------------------
__global__ __launch_bounds__(256) void vq_main_kernel(
    const float*  __restrict__ z,     const float* __restrict__ emb,
    const __bf16* __restrict__ Wbf,   const __bf16* __restrict__ embbf,
    const float*  __restrict__ embsq, float* __restrict__ out)
{
  __shared__ __align__(16) __bf16 lds_zT[2][NB * ZTS]; // double-buffered z chunk [n][k]
  __shared__ __align__(16) __bf16 lds_ze[NB * Q_];     // ze, column-major [n][q] bf16
  __shared__ __align__(16) float  lds_esq[K_];         // emb_sq
  __shared__ int lds_idx[NB];

  const int tid  = threadIdx.x;
  const int lane = tid & 31;
  const int wave = tid >> 5;        // 0..7
  const int kh   = lane >> 4;       // half-wave: 0 or 1
  const int l16  = lane & 15;

  const int b   = blockIdx.x >> 5;          // N_/NB == 32 blocks per batch
  const int nb0 = (blockIdx.x & 31) * NB;

  const float* zb = z + (size_t)b * CIN * N_;

  for (int i = tid; i < K_; i += 256) lds_esq[i] = embsq[i];

  // cooperative stage: z[b, kc..kc+31, nb0..nb0+127] -> LDS transposed bf16
  auto stage = [&](int kc, int buf) {
    for (int e = tid; e < 32 * NB; e += 256) {
      int i = e >> 7;           // k within chunk
      int n = e & (NB - 1);     // column
      lds_zT[buf][n * ZTS + i] = (__bf16)zb[(size_t)(kc + i) * N_ + nb0 + n];
    }
  };

  // ---------------- Phase 1: ze(128xNB) = W(128x768) x z_block(768xNB) ------
  v8f acc[8];
  #pragma unroll
  for (int j = 0; j < 8; ++j) acc[j] = (v8f){0,0,0,0,0,0,0,0};

  const int qr = wave * 16;   // this wave's Q rows

  stage(0, 0);
  __syncthreads();

  constexpr int NSTEPS = CIN / 32;          // 24 chunks
  for (int step = 0; step < NSTEPS; ++step) {
    const int kc  = step * 32;
    const int buf = step & 1;

    // stage next chunk into the other buffer (overlaps with WMMA below)
    if (step + 1 < NSTEPS) stage(kc + 32, buf ^ 1);

    // A fragment (16x32 bf16) from W row-major: two 8-elem chunks per lane
    V16U a;
    {
      const __bf16* wrow = Wbf + (size_t)(qr + l16) * CIN + kc + kh * 8;
      a.h[0] = *(const v8bf*)(wrow);
      a.h[1] = *(const v8bf*)(wrow + 16);
    }
    // preload all 8 B fragments; sched barrier keeps them live -> distinct
    // registers -> a single s_wait_dscnt before 8 back-to-back WMMAs
    V16U bfr[8];
    #pragma unroll
    for (int j = 0; j < 8; ++j) {
      const __bf16* p = &lds_zT[buf][(j * 16 + l16) * ZTS + kh * 8];
      bfr[j].h[0] = *(const v8bf*)(p);
      bfr[j].h[1] = *(const v8bf*)(p + 16);
    }
    SCHED_FENCE();
    #pragma unroll
    for (int j = 0; j < 8; ++j)
      acc[j] = __builtin_amdgcn_wmma_f32_16x16x32_bf16(
          false, a.v, false, bfr[j].v, (short)0, acc[j], false, false);
    SCHED_FENCE();

    __syncthreads();
  }

  // D layout: lane owns 8 consecutive q at column n -> one 16B packed store
  #pragma unroll
  for (int j = 0; j < 8; ++j) {
    int n  = j * 16 + l16;
    int qb = qr + kh * 8;
    v8bf pack;
    #pragma unroll
    for (int r = 0; r < 8; ++r) pack[r] = (__bf16)acc[j][r];
    *(v8bf*)&lds_ze[n * Q_ + qb] = pack;
  }
  __syncthreads();

  // ---------------- Phase 2: fused cross-GEMM + argmin over K --------------
  const int nloc = wave * 16 + l16;      // this lane's column (local)
  V16U bz[4];                            // resident B fragments: Q = 4 x K32
  #pragma unroll
  for (int c = 0; c < 4; ++c) {
    const __bf16* p = &lds_ze[nloc * Q_ + c * 32 + kh * 8];
    bz[c].h[0] = *(const v8bf*)(p);
    bz[c].h[1] = *(const v8bf*)(p + 16);
  }

  float bestv = 3.4e38f;
  int   besti = 0;

  // 32 emb rows (two 16-row tiles) per iteration: two independent WMMA chains
  for (int kt = 0; kt < K_; kt += 32) {
    if (kt + 32 < K_) {  // prefetch next 32-row tile (global_prefetch_b8)
      const __bf16* np = embbf + (size_t)(kt + 32 + lane) * Q_;
      __builtin_prefetch(np, 0, 1);
      __builtin_prefetch(np + 64, 0, 1);
    }

    const __bf16* r0 = embbf + (size_t)(kt + l16) * Q_;
    const __bf16* r1 = embbf + (size_t)(kt + 16 + l16) * Q_;
    V16U a0[4], a1[4];
    #pragma unroll
    for (int c = 0; c < 4; ++c) {
      const __bf16* p0 = r0 + c * 32 + kh * 8;
      const __bf16* p1 = r1 + c * 32 + kh * 8;
      a0[c].h[0] = *(const v8bf*)(p0);
      a0[c].h[1] = *(const v8bf*)(p0 + 16);
      a1[c].h[0] = *(const v8bf*)(p1);
      a1[c].h[1] = *(const v8bf*)(p1 + 16);
    }
    SCHED_FENCE();
    v8f c0 = (v8f){0,0,0,0,0,0,0,0};
    v8f c1 = (v8f){0,0,0,0,0,0,0,0};
    #pragma unroll
    for (int c = 0; c < 4; ++c) {
      c0 = __builtin_amdgcn_wmma_f32_16x16x32_bf16(
          false, a0[c].v, false, bz[c].v, (short)0, c0, false, false);
      c1 = __builtin_amdgcn_wmma_f32_16x16x32_bf16(
          false, a1[c].v, false, bz[c].v, (short)0, c1, false, false);
    }
    SCHED_FENCE();
    // lane's rows: chain0 -> kt+kh*8+r, chain1 -> kt+16+kh*8+r (D layout)
    const v4f* e0p = (const v4f*)&lds_esq[kt + kh * 8];
    const v4f* e1p = (const v4f*)&lds_esq[kt + 16 + kh * 8];
    v4f ea = e0p[0], eb = e0p[1], ec = e1p[0], ed = e1p[1];
    #pragma unroll
    for (int r = 0; r < 8; ++r) {
      float s = fmaf(-2.0f, c0[r], (r < 4 ? ea[r] : eb[r - 4]));
      if (s < bestv) { bestv = s; besti = kt + kh * 8 + r; }
    }
    #pragma unroll
    for (int r = 0; r < 8; ++r) {
      float s = fmaf(-2.0f, c1[r], (r < 4 ? ec[r] : ed[r - 4]));
      if (s < bestv) { bestv = s; besti = kt + 16 + kh * 8 + r; }
    }
  }

  // lanes L and L^16 hold disjoint row-halves of the same column: combine
  {
    float ov = __shfl_xor(bestv, 16, 32);
    int   oi = __shfl_xor(besti, 16, 32);
    if (ov < bestv || (ov == bestv && oi < besti)) { bestv = ov; besti = oi; }
  }
  if (lane < 16) lds_idx[nloc] = besti;
  __syncthreads();

  // ---------------- Phase 3: gather emb rows (f32) -> out (B, Q, N) --------
  float* ob = out + (size_t)b * Q_ * N_ + nb0;
  for (int e = tid; e < Q_ * NB; e += 256) {
    int q = e >> 7;          // NB == 128
    int n = e & (NB - 1);
    ob[(size_t)q * N_ + n] = emb[(size_t)lds_idx[n] * Q_ + q];
  }
}

// ---------------------------------------------------------------------------
extern "C" void kernel_launch(void* const* d_in, const int* in_sizes, int n_in,
                              void* d_out, int out_size, void* d_ws, size_t ws_size,
                              hipStream_t stream) {
  const float* z   = (const float*)d_in[0];   // (8, 768, 4096)
  const float* W   = (const float*)d_in[1];   // (128, 768)
  const float* emb = (const float*)d_in[2];   // (4096, 128)
  float* out = (float*)d_out;                 // (8, 128, 4096)

  char* ws = (char*)d_ws;
  float*  embsq = (float*)ws;                                   // 16 KB
  __bf16* embbf = (__bf16*)(ws + 16 * 1024);                    // 1 MB
  __bf16* Wbf   = (__bf16*)(ws + 16 * 1024 + (size_t)K_ * Q_ * 2); // 192 KB

  vq_prep_kernel<<<64, 256, 0, stream>>>(W, emb, Wbf, embbf, embsq);
  vq_main_kernel<<<B_ * (N_ / NB), 256, 0, stream>>>(z, emb, Wbf, embbf, embsq, out);
}
